// LSTMModel_10325101379766
// MI455X (gfx1250) — matrix operand order repaired and checked by reading, not compiled, not run on hardware
//
#include <hip/hip_runtime.h>

// ---------------------------------------------------------------------------
// Model constants (match reference)
// ---------------------------------------------------------------------------
constexpr int SEQ = 128, BATCH = 32;
constexpr int NTOKEN = 32000, NINP = 1024, NHID = 1024, NLAYERS = 2;
constexpr int SB = SEQ * BATCH;   // 4096 rows
constexpr int G4 = 4 * NHID;      // 4096 gate columns

// ---------------------------------------------------------------------------
// bf16 / WMMA plumbing
// ---------------------------------------------------------------------------
typedef __attribute__((ext_vector_type(16))) __bf16 v16bf;
typedef __attribute__((ext_vector_type(8)))  __bf16 v8bf;
typedef __attribute__((ext_vector_type(8)))  float  v8f;
typedef int v4i_vs __attribute__((vector_size(16)));  // GCC-vector int4 (builtin param type)

union Frag16 { v16bf v; v8bf h[2]; };

__device__ __forceinline__ v8bf ld8(const unsigned short* p) {
    return *(const v8bf*)p;
}
__device__ __forceinline__ void st8(unsigned short* p, v8bf x) {
    *(v8bf*)p = x;
}
__device__ __forceinline__ v8f wmma_bf16(v16bf a, v16bf b, v8f c) {
    // D = A(16x32) * B(32x16) + C(16x16), f32 accumulate
    return __builtin_amdgcn_wmma_f32_16x16x32_bf16(
        /*neg_a=*/false, a, /*neg_b=*/false, b,
        /*c_mod=*/(short)0, c, /*reuse_a=*/false, /*reuse_b=*/false);
}
__device__ __forceinline__ v8f zero8() {
    v8f z;
#pragma unroll
    for (int i = 0; i < 8; ++i) z[i] = 0.0f;
    return z;
}
__device__ __forceinline__ unsigned short f2bf(float f) {
    unsigned int u = __float_as_uint(f);
    unsigned int r = u + 0x7FFFu + ((u >> 16) & 1u);  // round-to-nearest-even
    return (unsigned short)(r >> 16);
}

// ---------------------------------------------------------------------------
// CDNA5 async global->LDS copy (ASYNCcnt-tracked), with safe fallback.
// ---------------------------------------------------------------------------
#if defined(__AMDGCN__) && __has_builtin(__builtin_amdgcn_global_load_async_to_lds_b128)
#define HAVE_ASYNC_LDS 1
#else
#define HAVE_ASYNC_LDS 0
#endif

#if HAVE_ASYNC_LDS
__device__ __forceinline__ void async_cp16(unsigned short* lds_dst,
                                           const unsigned short* gsrc) {
    __builtin_amdgcn_global_load_async_to_lds_b128(
        (__attribute__((address_space(1))) v4i_vs*)(unsigned short*)gsrc,
        (__attribute__((address_space(3))) v4i_vs*)lds_dst,
        /*imm offset=*/0, /*cpol=*/0);
}
__device__ __forceinline__ void wait_async_all() {
#if __has_builtin(__builtin_amdgcn_s_wait_asynccnt)
    __builtin_amdgcn_s_wait_asynccnt(0);
#else
    asm volatile("s_wait_asynccnt 0x0" ::: "memory");
#endif
}
#endif

// ---------------------------------------------------------------------------
// Elementwise helpers
// ---------------------------------------------------------------------------
__global__ void f32_to_bf16_kernel(const float* __restrict__ src,
                                   unsigned short* __restrict__ dst, size_t n) {
    size_t i = blockIdx.x * (size_t)blockDim.x + threadIdx.x;
    size_t stride = gridDim.x * (size_t)blockDim.x;
    for (; i < n; i += stride) dst[i] = f2bf(src[i]);
}

__global__ void zero_u32_kernel(unsigned int* __restrict__ p, size_t n) {
    size_t i = blockIdx.x * (size_t)blockDim.x + threadIdx.x;
    size_t stride = gridDim.x * (size_t)blockDim.x;
    for (; i < n; i += stride) p[i] = 0u;
}

__global__ void copy_f32_kernel(const float* __restrict__ src,
                                float* __restrict__ dst, size_t n) {
    size_t i = blockIdx.x * (size_t)blockDim.x + threadIdx.x;
    size_t stride = gridDim.x * (size_t)blockDim.x;
    for (; i < n; i += stride) dst[i] = src[i];
}

// Embedding gather + f32->bf16: one block per (s,b) row.
__global__ void embed_kernel(const int* __restrict__ tokens,
                             const float* __restrict__ encW,
                             unsigned short* __restrict__ x, int ninp) {
    int row = blockIdx.x;
    int tok = tokens[row];
    const float* src = encW + (size_t)tok * ninp;
    unsigned short* dst = x + (size_t)row * ninp;
    for (int i = threadIdx.x; i < ninp; i += blockDim.x) dst[i] = f2bf(src[i]);
}

// ---------------------------------------------------------------------------
// WMMA GEMM: C[M,N] = A[M,K](bf16) * Bw[N,K]^T(bf16) + bias0[n] + bias1[n]
// Block tile 128x128, BK=32, 256 threads = 8 waves (4 along M x 2 along N);
// each wave computes a 32x64 region = 2x4 16x16 WMMA tiles (8 WMMA / k-chunk).
// Requires M%128==0, N%128==0, K%32==0 (true for all call sites).
// ---------------------------------------------------------------------------
#define BM 128
#define BN 128
#define BK 32
#define SK 40    // LDS row stride (elements) with padding, keeps 16B alignment

__global__ __launch_bounds__(256) void gemm_bf16_kernel(
    const unsigned short* __restrict__ A,   // [M,K] bf16 bits
    const unsigned short* __restrict__ Bw,  // [N,K] bf16 bits (B transposed)
    float* __restrict__ C,                  // [M,N] f32
    const float* __restrict__ bias0,
    const float* __restrict__ bias1,
    int M, int N, int K) {
    __shared__ unsigned short As[BM * SK];
    __shared__ unsigned short Bs[BN * SK];

    const int tid   = threadIdx.x;
    const int lane  = tid & 31;
    const int wv    = tid >> 5;     // 0..7
    const int waveM = wv & 3;       // 0..3 -> 4*32 = 128 rows
    const int waveN = wv >> 2;      // 0..1 -> 2*64 = 128 cols
    const int half  = lane >> 4;    // 0/1
    const int lm    = lane & 15;
    const int m0 = blockIdx.y * BM;
    const int n0 = blockIdx.x * BN;

    v8f acc[2][4];
#pragma unroll
    for (int i = 0; i < 2; ++i)
#pragma unroll
        for (int j = 0; j < 4; ++j) acc[i][j] = zero8();

    for (int kt = 0; kt < K; kt += BK) {
        // stage A tile 128x32 and B tile 128x32: 512 16B chunks each,
        // 2 chunks per thread per tile.
#pragma unroll
        for (int it = 0; it < 2; ++it) {
            int q = tid + it * 256;
            int r = q >> 2, c8 = (q & 3) << 3;
#if HAVE_ASYNC_LDS
            async_cp16(&As[r * SK + c8], A + (size_t)(m0 + r) * K + kt + c8);
            async_cp16(&Bs[r * SK + c8], Bw + (size_t)(n0 + r) * K + kt + c8);
#else
            st8(&As[r * SK + c8], ld8(A + (size_t)(m0 + r) * K + kt + c8));
            st8(&Bs[r * SK + c8], ld8(Bw + (size_t)(n0 + r) * K + kt + c8));
#endif
        }
#if HAVE_ASYNC_LDS
        wait_async_all();
#endif
        __syncthreads();

        Frag16 af[2], bf[4];
#pragma unroll
        for (int i = 0; i < 2; ++i) {
            // A frag: lanes 0-15 K{0..7,16..23}, lanes 16-31 K{8..15,24..31}
            const unsigned short* ap =
                &As[(waveM * 32 + i * 16 + lm) * SK + half * 8];
            af[i].h[0] = ld8(ap);
            af[i].h[1] = ld8(ap + 16);
        }
#pragma unroll
        for (int j = 0; j < 4; ++j) {
            // B frag: lanes 0-15 K{0..7,16..23}, lanes 16-31 K{8..15,24..31}
            const unsigned short* bp =
                &Bs[(waveN * 64 + j * 16 + lm) * SK + half * 8];
            bf[j].h[0] = ld8(bp);
            bf[j].h[1] = ld8(bp + 16);
        }
#pragma unroll
        for (int i = 0; i < 2; ++i)
#pragma unroll
            for (int j = 0; j < 4; ++j)
                acc[i][j] = wmma_bf16(af[i].v, bf[j].v, acc[i][j]);
        __syncthreads();
    }

    // epilogue: C layout — VGPR r, lane: row = base + half*8 + r, col = lane&15
#pragma unroll
    for (int j = 0; j < 4; ++j) {
        int col = n0 + waveN * 64 + j * 16 + lm;
        float bv = 0.0f;
        if (bias0) bv += bias0[col];
        if (bias1) bv += bias1[col];
#pragma unroll
        for (int i = 0; i < 2; ++i) {
            int mbase = m0 + waveM * 32 + i * 16 + half * 8;
            float* cp = C + (size_t)mbase * N + col;
#pragma unroll
            for (int r = 0; r < 8; ++r)
                cp[(size_t)r * N] = acc[i][j][r] + bv;
        }
    }
}

// ---------------------------------------------------------------------------
// Fused LSTM recurrent step.
// Grid: NHID/16 workgroups x 256 threads (8 waves). WG owns hidden cols
// [jb, jb+16). 8 waves split K=1024 (128 each); WMMA accumulates the 4 gate
// tiles (i,f,g,o) x 2 M-tiles (B=32); f32 reduction through LDS; pointwise
// sigmoid/tanh update. h ping-pongs between buffers (h_in -> h_out).
// ---------------------------------------------------------------------------
__global__ __launch_bounds__(256) void lstm_step_kernel(
    const float* __restrict__ gx,            // [S*B, 4H]
    const unsigned short* __restrict__ Whh,  // [4H, H] bf16
    const unsigned short* __restrict__ h_in, // [32, H] bf16
    unsigned short* __restrict__ h_out,      // [32, H] bf16
    float* __restrict__ c,                   // [32, H] f32 (in-place)
    float* __restrict__ h_f32,               // [32, H] f32
    unsigned short* __restrict__ ys,         // [S*B, H] bf16 (layer output)
    int t, int H) {
    __shared__ float red[8 * 2048];          // 64 KB partials

    const int tid  = threadIdx.x;
    const int lane = tid & 31;
    const int wv   = tid >> 5;
    const int half = lane >> 4;
    const int lm   = lane & 15;
    const int jb   = blockIdx.x * 16;
    const int fourH = 4 * H;

    v8f acc[2][4];
#pragma unroll
    for (int i = 0; i < 2; ++i)
#pragma unroll
        for (int q = 0; q < 4; ++q) acc[i][q] = zero8();

    // each wave covers K in [wv*128, wv*128+128)
#pragma unroll
    for (int kk = 0; kk < 4; ++kk) {
        int kb = wv * 128 + kk * 32;
        Frag16 af[2];
#pragma unroll
        for (int i = 0; i < 2; ++i) {
            const unsigned short* ap =
                h_in + (size_t)(i * 16 + lm) * H + kb + half * 8;
            af[i].h[0] = ld8(ap);
            af[i].h[1] = ld8(ap + 16);
        }
#pragma unroll
        for (int q = 0; q < 4; ++q) {
            const unsigned short* bp =
                Whh + (size_t)(q * H + jb + lm) * H + kb + half * 8;
            Frag16 bfr;
            bfr.h[0] = ld8(bp);
            bfr.h[1] = ld8(bp + 16);
#pragma unroll
            for (int i = 0; i < 2; ++i)
                acc[i][q] = wmma_bf16(af[i].v, bfr.v, acc[i][q]);
        }
    }

    // dump per-wave partials to LDS
#pragma unroll
    for (int q = 0; q < 4; ++q)
#pragma unroll
        for (int i = 0; i < 2; ++i)
#pragma unroll
            for (int r = 0; r < 8; ++r)
                red[wv * 2048 + (q * 2 + i) * 256 + r * 32 + lane] =
                    acc[i][q][r];
    __syncthreads();

    // reduce + gate pointwise: 32 batch x 16 cols = 512 items, 2 per thread
#pragma unroll
    for (int rep = 0; rep < 2; ++rep) {
        int item = tid + rep * 256;
        int b = item >> 4, j = item & 15;
        int mt = b >> 4, mrel = b & 15;
        int ln = ((mrel >= 8) ? 16 : 0) + j;
        int r = mrel & 7;
        float g4[4];
#pragma unroll
        for (int q = 0; q < 4; ++q) {
            float s = 0.0f;
#pragma unroll
            for (int ww = 0; ww < 8; ++ww)
                s += red[ww * 2048 + (q * 2 + mt) * 256 + r * 32 + ln];
            g4[q] = s + gx[(size_t)(t * 32 + b) * fourH + q * H + jb + j];
        }
        float ig = 1.0f / (1.0f + __expf(-g4[0]));
        float fg = 1.0f / (1.0f + __expf(-g4[1]));
        float gg = tanhf(g4[2]);
        float og = 1.0f / (1.0f + __expf(-g4[3]));
        size_t ci = (size_t)b * H + jb + j;
        float cn = fg * c[ci] + ig * gg;
        float hn = og * tanhf(cn);
        c[ci] = cn;
        h_f32[ci] = hn;
        unsigned short hb = f2bf(hn);
        h_out[ci] = hb;
        ys[(size_t)(t * 32 + b) * H + jb + j] = hb;
    }
}

// ---------------------------------------------------------------------------
// In-place row-wise log-softmax: one block per row.
// ---------------------------------------------------------------------------
__global__ __launch_bounds__(256) void log_softmax_kernel(float* __restrict__ x,
                                                          int ncols) {
    __shared__ float sdata[256];
    float* row = x + (size_t)blockIdx.x * ncols;
    int tid = threadIdx.x;

    float m = -3.402823466e38f;
    for (int i = tid; i < ncols; i += 256) m = fmaxf(m, row[i]);
    sdata[tid] = m;
    __syncthreads();
    for (int s = 128; s > 0; s >>= 1) {
        if (tid < s) sdata[tid] = fmaxf(sdata[tid], sdata[tid + s]);
        __syncthreads();
    }
    float mx = sdata[0];
    __syncthreads();

    float sum = 0.0f;
    for (int i = tid; i < ncols; i += 256) sum += __expf(row[i] - mx);
    sdata[tid] = sum;
    __syncthreads();
    for (int s = 128; s > 0; s >>= 1) {
        if (tid < s) sdata[tid] += sdata[tid + s];
        __syncthreads();
    }
    float lse = mx + __logf(sdata[0]);
    __syncthreads();

    for (int i = tid; i < ncols; i += 256) row[i] = row[i] - lse;
}

// ---------------------------------------------------------------------------
// Host orchestration
// ---------------------------------------------------------------------------
extern "C" void kernel_launch(void* const* d_in, const int* in_sizes, int n_in,
                              void* d_out, int out_size, void* d_ws,
                              size_t ws_size, hipStream_t stream) {
    (void)in_sizes; (void)n_in; (void)out_size; (void)ws_size;

    const int*   tokens = (const int*)d_in[0];
    const float* encW   = (const float*)d_in[1];
    const float* WihAll = (const float*)d_in[2];  // [2, 4H, NINP]
    const float* WhhAll = (const float*)d_in[3];  // [2, 4H, NHID]
    const float* bIh    = (const float*)d_in[4];  // [2, 4H]
    const float* bHh    = (const float*)d_in[5];  // [2, 4H]
    const float* decW   = (const float*)d_in[6];  // [NTOKEN, NHID]
    const float* decB   = (const float*)d_in[7];  // [NTOKEN]

    float* out    = (float*)d_out;
    float* logits = out;                              // [SB, NTOKEN]
    float* out_h  = out + (size_t)SB * NTOKEN;        // [2, 32, NHID]
    float* out_c  = out_h + (size_t)NLAYERS * 32 * NHID;

    char* ws = (char*)d_ws;
    size_t off = 0;
    auto take = [&](size_t bytes) -> char* {
        char* p = ws + off;
        off = (off + bytes + 255) & ~(size_t)255;
        return p;
    };
    unsigned short* xA    = (unsigned short*)take((size_t)SB * NINP * 2);
    unsigned short* xB    = (unsigned short*)take((size_t)SB * NINP * 2);
    float*          gx    = (float*)take((size_t)SB * G4 * 4);
    unsigned short* WihB  = (unsigned short*)take((size_t)G4 * NINP * 2);
    unsigned short* WhhB  = (unsigned short*)take((size_t)G4 * NHID * 2);
    unsigned short* decWB = (unsigned short*)take((size_t)NTOKEN * NHID * 2);
    unsigned short* hbf   = (unsigned short*)take((size_t)2 * 32 * NHID * 2);
    float*          hf    = (float*)take((size_t)32 * NHID * 4);
    float*          cf    = (float*)take((size_t)32 * NHID * 4);

    // decoder weights -> bf16 (once per call)
    f32_to_bf16_kernel<<<4096, 256, 0, stream>>>(decW, decWB,
                                                 (size_t)NTOKEN * NHID);
    // embedding gather -> bf16 layer-0 input
    embed_kernel<<<SB, 256, 0, stream>>>(tokens, encW, xA, NINP);

    unsigned short* xin = xA;
    unsigned short* xout = xB;
    for (int l = 0; l < NLAYERS; ++l) {
        f32_to_bf16_kernel<<<2048, 256, 0, stream>>>(
            WihAll + (size_t)l * G4 * NINP, WihB, (size_t)G4 * NINP);
        f32_to_bf16_kernel<<<2048, 256, 0, stream>>>(
            WhhAll + (size_t)l * G4 * NHID, WhhB, (size_t)G4 * NHID);

        // gx = x @ Wih^T + b_ih + b_hh   [SB, 4H]
        gemm_bf16_kernel<<<dim3(G4 / BN, SB / BM), 256, 0, stream>>>(
            xin, WihB, gx, bIh + (size_t)l * G4, bHh + (size_t)l * G4,
            SB, G4, NINP);

        // h0 = c0 = 0
        zero_u32_kernel<<<64, 256, 0, stream>>>((unsigned int*)hbf,
                                                (size_t)32 * NHID / 2);
        zero_u32_kernel<<<64, 256, 0, stream>>>((unsigned int*)cf,
                                                (size_t)32 * NHID);

        for (int t = 0; t < SEQ; ++t) {
            unsigned short* hprev = hbf + (size_t)(t & 1) * 32 * NHID;
            unsigned short* hnext = hbf + (size_t)((t + 1) & 1) * 32 * NHID;
            lstm_step_kernel<<<NHID / 16, 256, 0, stream>>>(
                gx, WhhB, hprev, hnext, cf, hf, xout, t, NHID);
        }

        // final states -> output tail
        copy_f32_kernel<<<128, 256, 0, stream>>>(
            hf, out_h + (size_t)l * 32 * NHID, (size_t)32 * NHID);
        copy_f32_kernel<<<128, 256, 0, stream>>>(
            cf, out_c + (size_t)l * 32 * NHID, (size_t)32 * NHID);

        unsigned short* tmp = xin; xin = xout; xout = tmp;
    }

    // decoder: logits = ys @ dec_W^T + dec_b  (written straight into d_out)
    gemm_bf16_kernel<<<dim3(NTOKEN / BN, SB / BM), 256, 0, stream>>>(
        xin, decWB, logits, decB, nullptr, SB, NTOKEN, NHID);

    // in-place log-softmax per row
    log_softmax_kernel<<<SB, 256, 0, stream>>>(logits, NTOKEN);
}
